// LogicLayer_28681791603156
// MI455X (gfx1250) — matrix operand order repaired and checked by reading, not compiled, not run on hardware
//
#include <hip/hip_runtime.h>
#include <hip/hip_fp16.h>

// ---------------------------------------------------------------------------
// LogicLayer on gfx1250 (MI455X): groups 2/3 MLPs via v_wmma_f32_16x16x32_f16,
// reductions + tiny groups scalar.  Problem constants: B=4, n=48, C=16, H=64.
//
// Group 3 dominates (~11.8 GFLOP).  x3 (28 MB) is L2-resident (192 MB L2), so
// HBM traffic ~90 MB (~4 us at 23.3 TB/s); the GEMMs must run on the WMMA path.
// Each wave owns a 16-row M-tile; A fragments are gathered straight from
// global memory (per the ISA 16-bit A layout each lane needs exactly 8
// consecutive floats from x2 and 8 from x3 per 32-wide K-step -> 2x b128).
// Weights live in LDS as f16 (transposed, padded rows); each wave processes
// several tiles to amortize staging.  The t-outer/s-inner source form is
// deliberate: the scheduler unrolls it into s-major order with all four
// accumulators live and each A fragment loaded exactly once (verified in
// round-1 asm); explicit batching made it rematerialize A loads instead.
// ---------------------------------------------------------------------------

typedef _Float16 v16h __attribute__((ext_vector_type(16)));
typedef _Float16 v8h  __attribute__((ext_vector_type(8)));
typedef float    v8f  __attribute__((ext_vector_type(8)));

union AFrag {
    v16h     v;
    v8h      h[2];
    _Float16 e[16];
};

__device__ __forceinline__ void load8_g(AFrag& f, int h0, const float* __restrict__ p) {
    // 8 consecutive f32 -> halves [h0 .. h0+7]  (two global_load_b128)
    const float4* q = reinterpret_cast<const float4*>(p);
    float4 a = q[0];
    float4 b = q[1];
    f.e[h0 + 0] = (_Float16)a.x; f.e[h0 + 1] = (_Float16)a.y;
    f.e[h0 + 2] = (_Float16)a.z; f.e[h0 + 3] = (_Float16)a.w;
    f.e[h0 + 4] = (_Float16)b.x; f.e[h0 + 5] = (_Float16)b.y;
    f.e[h0 + 6] = (_Float16)b.z; f.e[h0 + 7] = (_Float16)b.w;
}

__device__ __forceinline__ v8f wmma16(const AFrag& A, const AFrag& B, v8f C) {
    return __builtin_amdgcn_wmma_f32_16x16x32_f16(
        /*neg_a=*/false, A.v, /*neg_b=*/false, B.v,
        /*c_mod=*/(short)0, C, /*reuse_a=*/false, /*reuse_b=*/false);
}

__device__ __forceinline__ float sigmoidf_(float x) { return 1.f / (1.f + __expf(-x)); }

// Compiler-only memory barrier.  The per-wave H tile round-trips through the
// wave's private LDS slice; the ISA guarantees same-wave LDS ops execute in
// order, so we only need to stop the compiler from hoisting the ds_loads
// above the ds_stores (cross-lane dependence is invisible to alias analysis).
__device__ __forceinline__ void lds_compiler_fence() {
    asm volatile("" ::: "memory");
}

// ---------------------------------------------------------------------------
// reduce_group(x3, 3): [B,48,48,48,16] -> [B,48,48,32] (exists/forall interleaved)
// ---------------------------------------------------------------------------
__global__ void lg_reduce3(const float* __restrict__ x3, float* __restrict__ R3) {
    int idx = blockIdx.x * blockDim.x + threadIdx.x;          // B*n*n*C = 147456
    if (idx >= 4 * 48 * 48 * 16) return;
    int c = idx & 15;
    int t = idx >> 4;
    int j = t % 48; t /= 48;
    int i = t % 48;
    int b = t / 48;
    const float* p = x3 + ((((long)b * 48 + i) * 48 + j) * 48) * 16 + c;
    float ex = 0.f, fa = 1.f;                                  // masked entries -> 0 / 1
    for (int k = 0; k < 48; ++k) {
        if (k != i && k != j) {
            float v = p[(long)k * 16];
            ex = fmaxf(ex, v);
            fa = fminf(fa, v);
        }
    }
    long o = (((long)b * 48 + i) * 48 + j) * 32 + 2 * c;
    R3[o] = ex; R3[o + 1] = fa;
}

// reduce_group(x2, 2): [B,48,48,16] -> [B,48,32]
__global__ void lg_reduce2(const float* __restrict__ x2, float* __restrict__ R2) {
    int idx = blockIdx.x * blockDim.x + threadIdx.x;          // B*n*C = 3072
    if (idx >= 4 * 48 * 16) return;
    int c = idx & 15;
    int t = idx >> 4;
    int i = t % 48;
    int b = t / 48;
    const float* p = x2 + (((long)b * 48 + i) * 48) * 16 + c;
    float ex = 0.f, fa = 1.f;
    for (int j = 0; j < 48; ++j) {
        if (j != i) {
            float v = p[(long)j * 16];
            ex = fmaxf(ex, v);
            fa = fminf(fa, v);
        }
    }
    long o = ((long)b * 48 + i) * 32 + 2 * c;
    R2[o] = ex; R2[o + 1] = fa;
}

// reduce_group(x1, 1): [B,48,16] -> [B,32]  (no self-exclusion for d=1)
__global__ void lg_reduce1(const float* __restrict__ x1, float* __restrict__ R1) {
    int idx = threadIdx.x;                                    // B*C = 64
    if (idx >= 4 * 16) return;
    int c = idx & 15;
    int b = idx >> 4;
    const float* p = x1 + ((long)b * 48) * 16 + c;
    float ex = -1e30f, fa = 1e30f;
    for (int i = 0; i < 48; ++i) {
        float v = p[(long)i * 16];
        ex = fmaxf(ex, v);
        fa = fminf(fa, v);
    }
    R1[b * 32 + 2 * c] = ex;
    R1[b * 32 + 2 * c + 1] = fa;
}

// ---------------------------------------------------------------------------
// group 0: rows = 4, Cin = 48 -> scalar
// ---------------------------------------------------------------------------
__global__ void lg0_kernel(const float* __restrict__ x0, const float* __restrict__ R1,
                           const float* __restrict__ W1, const float* __restrict__ b1,
                           const float* __restrict__ W2, const float* __restrict__ b2,
                           float* __restrict__ o0) {
    int b = threadIdx.x;
    if (b >= 4) return;
    float t0[48];
    for (int c = 0; c < 16; ++c) t0[c]      = x0[b * 16 + c];
    for (int c = 0; c < 32; ++c) t0[16 + c] = R1[b * 32 + c];
    float h[64];
    for (int hh = 0; hh < 64; ++hh) {
        float s = b1[hh];
        for (int c = 0; c < 48; ++c) s += t0[c] * W1[c * 64 + hh];
        h[hh] = s > 0.f ? s : 0.f;
    }
    for (int oc = 0; oc < 16; ++oc) {
        float s = b2[oc];
        for (int hh = 0; hh < 64; ++hh) s += h[hh] * W2[hh * 16 + oc];
        o0[b * 16 + oc] = sigmoidf_(s);
    }
}

// ---------------------------------------------------------------------------
// group 1: rows = 192, Cin = 64 -> scalar (negligible work)
// ---------------------------------------------------------------------------
__global__ void lg1_kernel(const float* __restrict__ x0, const float* __restrict__ x1,
                           const float* __restrict__ R2,
                           const float* __restrict__ W1, const float* __restrict__ b1,
                           const float* __restrict__ W2, const float* __restrict__ b2,
                           float* __restrict__ o1) {
    int row = blockIdx.x * blockDim.x + threadIdx.x;
    if (row >= 4 * 48) return;
    int i = row % 48, b = row / 48;
    float t1[64];
    for (int c = 0; c < 16; ++c) t1[c]      = x0[b * 16 + c];
    for (int c = 0; c < 16; ++c) t1[16 + c] = x1[((long)b * 48 + i) * 16 + c];
    for (int c = 0; c < 32; ++c) t1[32 + c] = R2[((long)b * 48 + i) * 32 + c];
    float h[64];
    for (int hh = 0; hh < 64; ++hh) {
        float s = b1[hh];
        for (int c = 0; c < 64; ++c) s += t1[c] * W1[c * 64 + hh];
        h[hh] = s > 0.f ? s : 0.f;
    }
    for (int oc = 0; oc < 16; ++oc) {
        float s = b2[oc];
        for (int hh = 0; hh < 64; ++hh) s += h[hh] * W2[hh * 16 + oc];
        o1[(long)row * 16 + oc] = sigmoidf_(s);
    }
}

// ---------------------------------------------------------------------------
// group 2: rows = B*n*n = 9216, Cin = 128 (2 perms x [x1|x2|R3]) — WMMA
// 8 waves/block, 2 tiles/wave; 36 blocks cover 576 tiles exactly.
// ---------------------------------------------------------------------------
__global__ __launch_bounds__(256) void lg2_kernel(
    const float* __restrict__ x1, const float* __restrict__ x2,
    const float* __restrict__ R3,
    const float* __restrict__ W1, const float* __restrict__ b1,
    const float* __restrict__ W2, const float* __restrict__ b2,
    float* __restrict__ o2) {
    constexpr int n = 48, C = 16, H = 64, CIN = 128;
    constexpr int WP = CIN + 8;   // 136 halves/row (272B: 16B aligned, banks rotate)
    constexpr int HP = H + 8;     // 72 halves/row (144B)
    constexpr int TILES = 2;

    __shared__ __align__(16) _Float16 sW1[H * WP];      // W1^T (f16)
    __shared__ __align__(16) _Float16 sW2[C * HP];      // W2^T (f16)
    __shared__ float sb1[H];
    __shared__ float sb2[C];
    __shared__ __align__(16) _Float16 sH[8 * 16 * HP];  // per-wave hidden tile

    for (int idx = threadIdx.x; idx < CIN * H; idx += 256) {
        int k = idx / H, no = idx - k * H;
        sW1[no * WP + k] = (_Float16)W1[idx];
    }
    for (int idx = threadIdx.x; idx < H * C; idx += 256) {
        int k = idx / C, nc = idx - k * C;
        sW2[nc * HP + k] = (_Float16)W2[idx];
    }
    if (threadIdx.x < H) sb1[threadIdx.x] = b1[threadIdx.x];
    if (threadIdx.x < C) sb2[threadIdx.x] = b2[threadIdx.x];
    __syncthreads();

    const int wave = threadIdx.x >> 5;
    const int lane = threadIdx.x & 31;
    const int l15  = lane & 15;         // A: row-in-tile / B,C,D: column
    const int kgrp = lane >> 4;
    const int kb   = kgrp * 8;          // A K-base per ISA 16-bit A layout

    _Float16* myH = &sH[wave * 16 * HP];

    // tile-invariant: biases + GEMM2 weight fragments
    float bias1v[4];
    #pragma unroll
    for (int t = 0; t < 4; ++t) bias1v[t] = sb1[t * 16 + l15];
    const float bias2 = sb2[l15];
    AFrag B2[2];
    #pragma unroll
    for (int s = 0; s < 2; ++s) {
        const _Float16* wp = &sW2[l15 * HP + s * 32 + kgrp * 8];
        B2[s].h[0] = *(const v8h*)(wp);
        B2[s].h[1] = *(const v8h*)(wp + 16);
    }

    const int tile0 = (blockIdx.x * 8 + wave) * TILES;

    #pragma unroll 1
    for (int it = 0; it < TILES; ++it) {
        const int row0 = (tile0 + it) * 16;
        int r = row0;
        const int j0 = r % n; r /= n;
        const int i  = r % n;
        const int b  = r / n;
        const int jr = j0 + l15;        // this lane's row -> j index

        // A fragments: 4 K-steps of 32 channels
        AFrag A[4];
        load8_g(A[0], 0, x1 + ((long)b * n + i) * C + kb);               // x1[b,i]
        load8_g(A[0], 8, x2 + (((long)b * n + i) * n + jr) * C + kb);    // x2[b,i,j]
        {
            const float* p = R3 + (((long)b * n + i) * n + jr) * 32;
            load8_g(A[1], 0, p + kb);
            load8_g(A[1], 8, p + 16 + kb);
        }
        load8_g(A[2], 0, x1 + ((long)b * n + jr) * C + kb);              // swapped perm
        load8_g(A[2], 8, x2 + (((long)b * n + jr) * n + i) * C + kb);
        {
            const float* p = R3 + (((long)b * n + jr) * n + i) * 32;
            load8_g(A[3], 0, p + kb);
            load8_g(A[3], 8, p + 16 + kb);
        }

        #pragma unroll
        for (int t = 0; t < 4; ++t) {
            v8f acc;
            #pragma unroll
            for (int e = 0; e < 8; ++e) acc[e] = bias1v[t];
            #pragma unroll
            for (int s = 0; s < 4; ++s) {
                AFrag Bf;
                const _Float16* wp = &sW1[(t * 16 + l15) * WP + s * 32 + kgrp * 8];
                Bf.h[0] = *(const v8h*)(wp);
                Bf.h[1] = *(const v8h*)(wp + 16);
                acc = wmma16(A[s], Bf, acc);
            }
            #pragma unroll
            for (int e = 0; e < 8; ++e) {
                float v = acc[e] > 0.f ? acc[e] : 0.f;
                myH[(e + kgrp * 8) * HP + t * 16 + l15] = (_Float16)v;   // C-layout -> [M][K]
            }
        }
        lds_compiler_fence();           // same-wave LDS is in-order in HW

        v8f acc2;
        #pragma unroll
        for (int e = 0; e < 8; ++e) acc2[e] = bias2;
        #pragma unroll
        for (int s = 0; s < 2; ++s) {
            AFrag A2;
            const _Float16* hp = &myH[l15 * HP + s * 32];
            A2.h[0] = *(const v8h*)(hp + kb);
            A2.h[1] = *(const v8h*)(hp + 16 + kb);
            acc2 = wmma16(A2, B2[s], acc2);
        }
        #pragma unroll
        for (int e = 0; e < 8; ++e) {
            o2[(long)(row0 + e + kgrp * 8) * C + l15] = sigmoidf_(acc2[e]);
        }
        lds_compiler_fence();           // next tile's H stores vs this tile's loads
    }
}

// ---------------------------------------------------------------------------
// group 3: rows = B*n^3 = 442368, Cin = 192 (6 perms x [x2|x3]) — WMMA
// 8 waves/block, 8 tiles/wave; 432 blocks cover 27648 tiles exactly.
// ---------------------------------------------------------------------------
__global__ __launch_bounds__(256) void lg3_kernel(
    const float* __restrict__ x2, const float* __restrict__ x3,
    const float* __restrict__ W1, const float* __restrict__ b1,
    const float* __restrict__ W2, const float* __restrict__ b2,
    float* __restrict__ o3) {
    constexpr int n = 48, C = 16, H = 64, CIN = 192;
    constexpr int WP = CIN + 8;   // 200 halves/row (400B)
    constexpr int HP = H + 8;     // 72 halves/row
    constexpr int TILES = 8;

    __shared__ __align__(16) _Float16 sW1[H * WP];
    __shared__ __align__(16) _Float16 sW2[C * HP];
    __shared__ float sb1[H];
    __shared__ float sb2[C];
    __shared__ __align__(16) _Float16 sH[8 * 16 * HP];

    for (int idx = threadIdx.x; idx < CIN * H; idx += 256) {
        int k = idx / H, no = idx - k * H;
        sW1[no * WP + k] = (_Float16)W1[idx];
    }
    for (int idx = threadIdx.x; idx < H * C; idx += 256) {
        int k = idx / C, nc = idx - k * C;
        sW2[nc * HP + k] = (_Float16)W2[idx];
    }
    if (threadIdx.x < H) sb1[threadIdx.x] = b1[threadIdx.x];
    if (threadIdx.x < C) sb2[threadIdx.x] = b2[threadIdx.x];
    __syncthreads();

    const int wave = threadIdx.x >> 5;
    const int lane = threadIdx.x & 31;
    const int l15  = lane & 15;
    const int kgrp = lane >> 4;
    const int kb   = kgrp * 8;

    _Float16* myH = &sH[wave * 16 * HP];

    // tile-invariant: biases + GEMM2 weight fragments
    float bias1v[4];
    #pragma unroll
    for (int t = 0; t < 4; ++t) bias1v[t] = sb1[t * 16 + l15];
    const float bias2 = sb2[l15];
    AFrag B2[2];
    #pragma unroll
    for (int s = 0; s < 2; ++s) {
        const _Float16* wp = &sW2[l15 * HP + s * 32 + kgrp * 8];
        B2[s].h[0] = *(const v8h*)(wp);
        B2[s].h[1] = *(const v8h*)(wp + 16);
    }

    const int tile0 = (blockIdx.x * 8 + wave) * TILES;

    #pragma unroll 1
    for (int it = 0; it < TILES; ++it) {
        const int row0 = (tile0 + it) * 16;     // 16 consecutive k: same (b,i,j)
        int r = row0;
        const int k0 = r % n; r /= n;
        const int j  = r % n; r /= n;
        const int i  = r % n;
        const int b  = r / n;
        const int kk = k0 + l15;                // this lane's k index

        // 6 perms of (i,j,k); each 32-ch K-step = [x2[p1,p2] | x3[p1,p2,p3]]
        const int P1[6] = { i,  i,  j,  j,  kk, kk };
        const int P2[6] = { j,  kk, i,  kk, i,  j  };
        const int P3[6] = { kk, j,  kk, i,  j,  i  };

        AFrag A[6];
        #pragma unroll
        for (int s = 0; s < 6; ++s) {
            long o2i = (((long)b * n + P1[s]) * n + P2[s]) * C + kb;
            long o3i = ((((long)b * n + P1[s]) * n + P2[s]) * n + P3[s]) * C + kb;
            load8_g(A[s], 0, x2 + o2i);
            load8_g(A[s], 8, x3 + o3i);
        }

        #pragma unroll
        for (int t = 0; t < 4; ++t) {
            v8f acc;
            #pragma unroll
            for (int e = 0; e < 8; ++e) acc[e] = bias1v[t];
            #pragma unroll
            for (int s = 0; s < 6; ++s) {
                AFrag Bf;
                const _Float16* wp = &sW1[(t * 16 + l15) * WP + s * 32 + kgrp * 8];
                Bf.h[0] = *(const v8h*)(wp);
                Bf.h[1] = *(const v8h*)(wp + 16);
                acc = wmma16(A[s], Bf, acc);
            }
            #pragma unroll
            for (int e = 0; e < 8; ++e) {
                float v = acc[e] > 0.f ? acc[e] : 0.f;
                myH[(e + kgrp * 8) * HP + t * 16 + l15] = (_Float16)v;
            }
        }
        lds_compiler_fence();

        v8f acc2;
        #pragma unroll
        for (int e = 0; e < 8; ++e) acc2[e] = bias2;
        #pragma unroll
        for (int s = 0; s < 2; ++s) {
            AFrag A2;
            const _Float16* hp = &myH[l15 * HP + s * 32];
            A2.h[0] = *(const v8h*)(hp + kb);
            A2.h[1] = *(const v8h*)(hp + 16 + kb);
            acc2 = wmma16(A2, B2[s], acc2);
        }
        #pragma unroll
        for (int e = 0; e < 8; ++e) {
            o3[(long)(row0 + e + kgrp * 8) * C + l15] = sigmoidf_(acc2[e]);
        }
        lds_compiler_fence();
    }
}

// ---------------------------------------------------------------------------
extern "C" void kernel_launch(void* const* d_in, const int* in_sizes, int n_in,
                              void* d_out, int out_size, void* d_ws, size_t ws_size,
                              hipStream_t stream) {
    const float* x0 = (const float*)d_in[0];
    const float* x1 = (const float*)d_in[1];
    const float* x2 = (const float*)d_in[2];
    const float* x3 = (const float*)d_in[3];
    const float* W1_0 = (const float*)d_in[4];
    const float* b1_0 = (const float*)d_in[5];
    const float* W2_0 = (const float*)d_in[6];
    const float* b2_0 = (const float*)d_in[7];
    const float* W1_1 = (const float*)d_in[8];
    const float* b1_1 = (const float*)d_in[9];
    const float* W2_1 = (const float*)d_in[10];
    const float* b2_1 = (const float*)d_in[11];
    const float* W1_2 = (const float*)d_in[12];
    const float* b1_2 = (const float*)d_in[13];
    const float* W2_2 = (const float*)d_in[14];
    const float* b2_2 = (const float*)d_in[15];
    const float* W1_3 = (const float*)d_in[16];
    const float* b1_3 = (const float*)d_in[17];
    const float* W2_3 = (const float*)d_in[18];
    const float* b2_3 = (const float*)d_in[19];

    // workspace: R3 [4,48,48,32], R2 [4,48,32], R1 [4,32]
    float* R3 = (float*)d_ws;
    float* R2 = R3 + 4 * 48 * 48 * 32;
    float* R1 = R2 + 4 * 48 * 32;

    // outputs concatenated flat: o0 (64), o1 (3072), o2 (147456), o3 (7077888)
    float* o0 = (float*)d_out;
    float* o1 = o0 + 4 * 16;
    float* o2 = o1 + 4 * 48 * 16;
    float* o3 = o2 + 4 * 48 * 48 * 16;

    lg_reduce3<<<(4 * 48 * 48 * 16 + 255) / 256, 256, 0, stream>>>(x3, R3);
    lg_reduce2<<<(4 * 48 * 16 + 255) / 256, 256, 0, stream>>>(x2, R2);
    lg_reduce1<<<1, 64, 0, stream>>>(x1, R1);

    lg0_kernel<<<1, 32, 0, stream>>>(x0, R1, W1_0, b1_0, W2_0, b2_0, o0);
    lg1_kernel<<<1, 192, 0, stream>>>(x0, x1, R2, W1_1, b1_1, W2_1, b2_1, o1);
    lg2_kernel<<<36, 256, 0, stream>>>(x1, x2, R3, W1_2, b1_2, W2_2, b2_2, o2);
    lg3_kernel<<<432, 256, 0, stream>>>(x2, x3, W1_3, b1_3, W2_3, b2_3, o3);
}
